// Ligand_GAT_18580028522895
// MI455X (gfx1250) — matrix operand order repaired and checked by reading, not compiled, not run on hardware
//
#include <hip/hip_runtime.h>
#include <cstdint>
#include <cstddef>

// ---------------- problem constants (match reference) ----------------
#define NN      20000
#define EE      200000
#define CCH     64      // C
#define HHD     4       // heads
#define HCC     256     // H*C
#define LLY     2       // layers
#define L0IN    65
#define EIN     15
#define L0OUT   64
#define EPSV    1e-5f
#define NSLOPE  0.01f
#define NCAT    768     // 3*HCC concatenated node projections

typedef unsigned short bfraw; // raw bf16 bits in memory
typedef __attribute__((ext_vector_type(16))) __bf16 v16bf;
typedef __attribute__((ext_vector_type(8)))  float  v8f;

// ---------------- small device helpers ----------------
__device__ __forceinline__ bfraw f2bf(float f) {
  unsigned u = __float_as_uint(f);
  u += 0x7FFFu + ((u >> 16) & 1u);          // round-to-nearest-even
  return (bfraw)(u >> 16);
}
__device__ __forceinline__ float bf2f(bfraw h) {
  return __uint_as_float(((unsigned)h) << 16);
}
__device__ __forceinline__ float warp_sum32(float v) {
#pragma unroll
  for (int o = 16; o > 0; o >>= 1) v += __shfl_xor(v, o, 32);
  return v;
}
__device__ __forceinline__ void atomic_add_f32(float* p, float v) {
  __hip_atomic_fetch_add(p, v, __ATOMIC_RELAXED, __HIP_MEMORY_SCOPE_AGENT);
}
// IEEE-monotonic float atomic max (works with -3e38 init)
__device__ __forceinline__ void atomic_max_f32(float* p, float v) {
  if (v >= 0.0f) atomicMax((int*)p, __float_as_int(v));
  else           atomicMin((unsigned int*)p, (unsigned int)__float_as_int(v));
}

// ---------------- utility kernels ----------------
__global__ void fill_kernel(float* __restrict__ p, float v, int n) {
  int i = blockIdx.x * blockDim.x + threadIdx.x;
  if (i < n) p[i] = v;
}

// Pack weight W[64, ncols_src] (f32, row-major) into WMMA B-fragment order (bf16):
//   out[((nt*2 + ks)*32 + lane)*16 + e] = bf16( W[ks*32 + (lane/16)*16 + e][nt*16 + lane%16] )
// so a wave reads its 32x16 B fragment as two contiguous b128 loads.
__global__ void pack_weight_kernel(const float* __restrict__ W, bfraw* __restrict__ out,
                                   int ncols_src) {
  const int n = 64 * ncols_src;
  int i = blockIdx.x * blockDim.x + threadIdx.x;
  if (i >= n) return;
  const int e    = i & 15;
  const int lane = (i >> 4) & 31;
  const int ks   = (i >> 9) & 1;
  const int nt   = i >> 10;
  const int tg = lane >> 4, tl = lane & 15;
  const int k = ks * 32 + tg * 16 + e;
  const int c = nt * 16 + tl;
  out[i] = f2bf(W[(size_t)k * ncols_src + c]);
}

// catbias[0:512)=0 ; catbias[512:768) = bnode_l
__global__ void build_catbias_kernel(const float* __restrict__ bnode_l, float* __restrict__ out) {
  int i = blockIdx.x * blockDim.x + threadIdx.x;
  if (i < NCAT) out[i] = (i < 2 * HCC) ? 0.0f : bnode_l[i - 2 * HCC];
}

// out[row, c] = sum_k X[row,k]*W[k,c] + b[c], stored bf16. Block=64 threads, grid=M.
__global__ void in_proj_kernel(const float* __restrict__ X, const float* __restrict__ W,
                               const float* __restrict__ bias, bfraw* __restrict__ out, int K) {
  const int row = blockIdx.x, c = threadIdx.x;
  const float* xr = X + (size_t)row * K;
  float acc = bias[c];
  for (int k = 0; k < K; ++k) acc = fmaf(xr[k], W[(size_t)k * CCH + c], acc);
  out[(size_t)row * CCH + c] = f2bf(acc);
}

// ---------------- WMMA GEMM: C[M,ncols] = A[M,64](bf16) @ Bpacked(64 x ncols) + bias ----------
// block = 128 threads (4 waves); wave w owns n-tile nt = blockIdx.y*4 + w (16 cols),
// and 4 M-subtiles of 16 rows. B fragments come from the packed layout: 2 b128 loads.
union FragAB { uint4 u[2]; bfraw s[16]; v16bf v; };

template <bool STORE_BF16>
__global__ __launch_bounds__(128)
void wmma_gemm_kernel(const bfraw* __restrict__ A, const bfraw* __restrict__ Bp,
                      const float* __restrict__ bias, void* __restrict__ Cptr,
                      int M, int ncols) {
  const int lane = threadIdx.x & 31;
  const int wave = threadIdx.x >> 5;
  const int tg   = lane >> 4;   // 0/1 half-wave group
  const int tl   = lane & 15;
  const int nt   = blockIdx.y * 4 + wave;      // n-tile of 16 cols
  const int n0   = nt * 16;
  const int m0   = blockIdx.x * 64;

  v8f acc[4] = {};

#pragma unroll
  for (int ks = 0; ks < 2; ++ks) {             // K = 64, two steps of 32
    const int k0 = ks * 32;
    // packed B fragment: 16 contiguous bf16 per lane -> two b128 loads
    const bfraw* bp = Bp + ((size_t)((nt * 2 + ks) * 32 + lane)) * 16;
    FragAB bfrag;
    bfrag.u[0] = *(const uint4*)(bp);
    bfrag.u[1] = *(const uint4*)(bp + 8);

#pragma unroll
    for (int mt = 0; mt < 4; ++mt) {
      int row = m0 + mt * 16 + tl;
      row = row < M ? row : M - 1;             // clamp (stores are guarded)
      const bfraw* ar = A + (size_t)row * 64 + k0 + tg * 16;
      FragAB afrag;                            // slots 0..15: K=k0+tg*16+e
      afrag.u[0] = *(const uint4*)(ar);
      afrag.u[1] = *(const uint4*)(ar + 8);
      acc[mt] = __builtin_amdgcn_wmma_f32_16x16x32_bf16(
          false, afrag.v, false, bfrag.v, (short)0, acc[mt], false, false);
    }
  }

  const float bcol = bias ? bias[n0 + tl] : 0.0f;
#pragma unroll
  for (int mt = 0; mt < 4; ++mt) {
#pragma unroll
    for (int r = 0; r < 8; ++r) {
      const int row = m0 + mt * 16 + r + tg * 8;   // C layout: VGPR r -> M=r / r+8
      if (row < M) {
        const float v = acc[mt][r] + bcol;
        if constexpr (STORE_BF16)
          ((bfraw*)Cptr)[(size_t)row * ncols + n0 + tl] = f2bf(v);
        else
          ((float*)Cptr)[(size_t)row * ncols + n0 + tl] = v;
      }
    }
  }
}

// ---------------- edge assemble + leaky_relu + attention score + segment max ----------------
// warp per edge, lane owns channels {lane, lane+32} of each head.
// hcat[N, 768] = [ h@Wni | h@Wnj | h@Wnode + bnode ]
__global__ void edge_assemble_score_kernel(bfraw* __restrict__ fout,
                                           const float* __restrict__ hcat,
                                           const float* __restrict__ biase, const float* __restrict__ attn,
                                           const int* __restrict__ src, const int* __restrict__ dst,
                                           float* __restrict__ score, float* __restrict__ smax) {
  const int warp = threadIdx.x >> 5, lane = threadIdx.x & 31;
  const long e = (long)blockIdx.x * 8 + warp;
  if (e >= EE) return;
  const int s = src[e], d = dst[e];
  const float* hs = hcat + (size_t)s * NCAT;          // Wni slice
  const float* hd = hcat + (size_t)d * NCAT + HCC;    // Wnj slice
  bfraw* fo = fout + (size_t)e * HCC;
  float part[HHD];
#pragma unroll
  for (int h = 0; h < HHD; ++h) {
    const int i0 = h * CCH + lane, i1 = i0 + 32;
    float v0 = bf2f(fo[i0]) + hs[i0] + hd[i0] + biase[i0];
    float v1 = bf2f(fo[i1]) + hs[i1] + hd[i1] + biase[i1];
    fo[i0] = f2bf(v0); fo[i1] = f2bf(v1);             // pre-activation f_out (used by f-update)
    const float l0 = v0 > 0.f ? v0 : NSLOPE * v0;
    const float l1 = v1 > 0.f ? v1 : NSLOPE * v1;
    part[h] = fmaf(l0, attn[i0], l1 * attn[i1]);
  }
#pragma unroll
  for (int h = 0; h < HHD; ++h) part[h] = warp_sum32(part[h]);
  if (lane == 0) {
#pragma unroll
    for (int h = 0; h < HHD; ++h) {
      score[e * HHD + h] = part[h];
      atomic_max_f32(&smax[(size_t)d * HHD + h], part[h]);
    }
  }
}

// ---------------- exp(score - max) + segment sum ----------------
__global__ void edge_softmax_kernel(const float* __restrict__ score, const float* __restrict__ smax,
                                    const int* __restrict__ dst, float* __restrict__ wexp,
                                    float* __restrict__ denom) {
  const long i = (long)blockIdx.x * blockDim.x + threadIdx.x;
  if (i >= (long)EE * HHD) return;
  const int e = (int)(i >> 2), h = (int)(i & 3);
  const int d = dst[e];
  const float w = __expf(score[i] - smax[(size_t)d * HHD + h]);
  wexp[i] = w;
  atomic_add_f32(&denom[(size_t)d * HHD + h], w);
}

// ---------------- weighted message scatter-aggregate ----------------
__global__ void edge_aggregate_kernel(const float* __restrict__ hcat, const float* __restrict__ wexp,
                                      const float* __restrict__ denom, const int* __restrict__ src,
                                      const int* __restrict__ dst, float* __restrict__ hnew) {
  const int warp = threadIdx.x >> 5, lane = threadIdx.x & 31;
  const long e = (long)blockIdx.x * 8 + warp;
  if (e >= EE) return;
  const int s = src[e], d = dst[e];
  const float* hp = hcat + (size_t)s * NCAT + 2 * HCC;   // Wnode slice (hproj)
  float* hn = hnew + (size_t)d * HCC;
#pragma unroll
  for (int h = 0; h < HHD; ++h) {
    const float a = wexp[e * HHD + h] / denom[(size_t)d * HHD + h];
    const int i0 = h * CCH + lane, i1 = i0 + 32;
    atomic_add_f32(hn + i0, hp[i0] * a);
    atomic_add_f32(hn + i1, hp[i1] * a);
  }
}

// ---------------- head-mean + InstanceNorm + ELU (nodes) ----------------
__global__ void node_update_kernel(const float* __restrict__ hnew, bfraw* __restrict__ hb) {
  const int warp = threadIdx.x >> 5, lane = threadIdx.x & 31;
  const int n = blockIdx.x * 8 + warp;
  if (n >= NN) return;
  const float* hn = hnew + (size_t)n * HCC;
  const int c0 = lane, c1 = lane + 32;
  const float v0 = 0.25f * (hn[c0] + hn[CCH + c0] + hn[2 * CCH + c0] + hn[3 * CCH + c0]);
  const float v1 = 0.25f * (hn[c1] + hn[CCH + c1] + hn[2 * CCH + c1] + hn[3 * CCH + c1]);
  const float s  = warp_sum32(v0 + v1);
  const float ss = warp_sum32(v0 * v0 + v1 * v1);
  const float m   = s * (1.0f / CCH);
  const float var = ss * (1.0f / CCH) - m * m;
  const float inv = rsqrtf(var + EPSV);
  float y0 = (v0 - m) * inv; y0 = y0 > 0.f ? y0 : __expf(y0) - 1.f;
  float y1 = (v1 - m) * inv; y1 = y1 > 0.f ? y1 : __expf(y1) - 1.f;
  hb[(size_t)n * CCH + c0] = f2bf(y0);
  hb[(size_t)n * CCH + c1] = f2bf(y1);
}

// ---------------- head-mean + InstanceNorm + ELU (edges, bf16 source) ----------------
__global__ void edge_feat_update_kernel(const bfraw* __restrict__ fout, bfraw* __restrict__ fb) {
  const int warp = threadIdx.x >> 5, lane = threadIdx.x & 31;
  const long e = (long)blockIdx.x * 8 + warp;
  if (e >= EE) return;
  const bfraw* fo = fout + (size_t)e * HCC;
  const int c0 = lane, c1 = lane + 32;
  const float v0 = 0.25f * (bf2f(fo[c0]) + bf2f(fo[CCH + c0]) + bf2f(fo[2 * CCH + c0]) + bf2f(fo[3 * CCH + c0]));
  const float v1 = 0.25f * (bf2f(fo[c1]) + bf2f(fo[CCH + c1]) + bf2f(fo[2 * CCH + c1]) + bf2f(fo[3 * CCH + c1]));
  const float s  = warp_sum32(v0 + v1);
  const float ss = warp_sum32(v0 * v0 + v1 * v1);
  const float m   = s * (1.0f / CCH);
  const float var = ss * (1.0f / CCH) - m * m;
  const float inv = rsqrtf(var + EPSV);
  float y0 = (v0 - m) * inv; y0 = y0 > 0.f ? y0 : __expf(y0) - 1.f;
  float y1 = (v1 - m) * inv; y1 = y1 > 0.f ? y1 : __expf(y1) - 1.f;
  fb[(size_t)e * CCH + c0] = f2bf(y0);
  fb[(size_t)e * CCH + c1] = f2bf(y1);
}

// ---------------- host orchestration ----------------
extern "C" void kernel_launch(void* const* d_in, const int* in_sizes, int n_in,
                              void* d_out, int out_size, void* d_ws, size_t ws_size,
                              hipStream_t stream) {
  (void)in_sizes; (void)n_in; (void)out_size; (void)ws_size;
  const float* x      = (const float*)d_in[0];
  const float* efeat  = (const float*)d_in[1];
  const int*   src    = (const int*)d_in[2];
  const int*   dst    = (const int*)d_in[3];
  const float* Wn0    = (const float*)d_in[4];
  const float* bn0    = (const float*)d_in[5];
  const float* We0    = (const float*)d_in[6];
  const float* be0    = (const float*)d_in[7];
  const float* Wnode  = (const float*)d_in[8];
  const float* bnode  = (const float*)d_in[9];
  const float* Wni    = (const float*)d_in[10];
  const float* Wnj    = (const float*)d_in[11];
  const float* Wfij   = (const float*)d_in[12];
  const float* attn   = (const float*)d_in[13];
  const float* bias_e = (const float*)d_in[14];
  const float* Wf     = (const float*)d_in[15];
  const float* bfv    = (const float*)d_in[16];

  // workspace carve (256B aligned)
  size_t off = 0;
  char* wsb = (char*)d_ws;
  auto carve = [&](size_t bytes) -> void* {
    void* p = (void*)(wsb + off);
    off += (bytes + 255) & ~(size_t)255;
    return p;
  };
  const size_t WSLICE = (size_t)CCH * HCC;           // 64*256 elements per weight per layer
  bfraw* hb      = (bfraw*)carve((size_t)NN * CCH * 2);
  bfraw* fb      = (bfraw*)carve((size_t)EE * CCH * 2);
  bfraw* WcatP   = (bfraw*)carve((size_t)LLY * 3 * WSLICE * 2);  // per layer: [Wni|Wnj|Wnode] packed
  bfraw* WfijP   = (bfraw*)carve((size_t)LLY * WSLICE * 2);
  bfraw* WfP     = (bfraw*)carve((size_t)CCH * L0OUT * 2);
  float* catbias = (float*)carve((size_t)NCAT * 4);
  float* hcat    = (float*)carve((size_t)NN * NCAT * 4);
  bfraw* foutB   = (bfraw*)carve((size_t)EE * HCC * 2);
  float* score   = (float*)carve((size_t)EE * HHD * 4);
  float* wexp    = (float*)carve((size_t)EE * HHD * 4);
  float* smax    = (float*)carve((size_t)NN * HHD * 4);
  float* denom   = (float*)carve((size_t)NN * HHD * 4);
  float* hnew    = (float*)carve((size_t)NN * HCC * 4);

  // weight packing (f32 -> bf16, WMMA B-fragment order)
  const int wgrid = (int)((WSLICE + 255) / 256);
  for (int l = 0; l < LLY; ++l) {
    bfraw* cat_l = WcatP + (size_t)l * 3 * WSLICE;
    pack_weight_kernel<<<wgrid, 256, 0, stream>>>(Wni   + (size_t)l * WSLICE, cat_l,              HCC);
    pack_weight_kernel<<<wgrid, 256, 0, stream>>>(Wnj   + (size_t)l * WSLICE, cat_l + WSLICE,     HCC);
    pack_weight_kernel<<<wgrid, 256, 0, stream>>>(Wnode + (size_t)l * WSLICE, cat_l + 2 * WSLICE, HCC);
    pack_weight_kernel<<<wgrid, 256, 0, stream>>>(Wfij  + (size_t)l * WSLICE, WfijP + (size_t)l * WSLICE, HCC);
  }
  pack_weight_kernel<<<(CCH * L0OUT + 255) / 256, 256, 0, stream>>>(Wf, WfP, L0OUT);

  // input projections
  in_proj_kernel<<<NN, CCH, 0, stream>>>(x, Wn0, bn0, hb, L0IN);
  in_proj_kernel<<<EE, CCH, 0, stream>>>(efeat, We0, be0, fb, EIN);

  const dim3 gemmBlk(128);
  for (int l = 0; l < LLY; ++l) {
    const bfraw* cat_l   = WcatP + (size_t)l * 3 * WSLICE;
    const bfraw* wfij_l  = WfijP + (size_t)l * WSLICE;
    const float* bnode_l = bnode  + (size_t)l * HCC;
    const float* attn_l  = attn   + (size_t)l * HHD * CCH;
    const float* biase_l = bias_e + (size_t)l * HCC;

    build_catbias_kernel<<<(NCAT + 255) / 256, 256, 0, stream>>>(bnode_l, catbias);

    // fused node GEMM: hcat[N,768] = hb @ [Wni|Wnj|Wnode] + [0|0|bnode]
    const dim3 gN((NN + 63) / 64, NCAT / 64);
    wmma_gemm_kernel<false><<<gN, gemmBlk, 0, stream>>>(hb, cat_l, catbias, hcat, NN, NCAT);
    // edge GEMM: fout = fb @ Wfij (bf16 store)
    const dim3 gE(EE / 64, HCC / 64);
    wmma_gemm_kernel<true ><<<gE, gemmBlk, 0, stream>>>(fb, wfij_l, nullptr, foutB, EE, HCC);

    fill_kernel<<<(NN * HHD + 255) / 256, 256, 0, stream>>>(smax, -3.0e38f, NN * HHD);
    fill_kernel<<<(NN * HHD + 255) / 256, 256, 0, stream>>>(denom, 0.0f, NN * HHD);
    fill_kernel<<<(NN * HCC + 255) / 256, 256, 0, stream>>>(hnew, 0.0f, NN * HCC);

    edge_assemble_score_kernel<<<EE / 8, 256, 0, stream>>>(foutB, hcat, biase_l, attn_l,
                                                           src, dst, score, smax);
    edge_softmax_kernel<<<(EE * HHD + 255) / 256, 256, 0, stream>>>(score, smax, dst, wexp, denom);
    edge_aggregate_kernel<<<EE / 8, 256, 0, stream>>>(hcat, wexp, denom, src, dst, hnew);
    node_update_kernel<<<(NN + 7) / 8, 256, 0, stream>>>(hnew, hb);
    if (l == 0)
      edge_feat_update_kernel<<<EE / 8, 256, 0, stream>>>(foutB, fb);
  }

  // final projection: out = h @ Wf + bf  ([N,64] f32)
  const dim3 gF((NN + 63) / 64, 1);
  wmma_gemm_kernel<false><<<gF, gemmBlk, 0, stream>>>(hb, WfP, bfv, (float*)d_out, NN, L0OUT);
}
